// MultiHeadAttention_30906584662031
// MI455X (gfx1250) — matrix operand order repaired and checked by reading, not compiled, not run on hardware
//
#include <hip/hip_runtime.h>

#define D_MODEL 1024
#define NUM_HEADS 16
#define D_K 64
#define BATCH 2
#define SEQ 2048
#define SCALE 0.125f  // 1/sqrt(64)

typedef __attribute__((ext_vector_type(16))) __bf16 v16bf;
typedef __attribute__((ext_vector_type(8)))  float  v8f;
typedef __attribute__((ext_vector_type(4)))  unsigned int u32x4;

union Frag16 { v16bf v; u32x4 u[2]; };

__device__ __forceinline__ unsigned short f32_to_bf16(float f) {
  unsigned int u = __float_as_uint(f);
  unsigned int r = u + 0x7FFFu + ((u >> 16) & 1u);
  return (unsigned short)(r >> 16);
}

__device__ __forceinline__ v8f wmma_bf16(v16bf a, v16bf b, v8f c) {
  return __builtin_amdgcn_wmma_f32_16x16x32_bf16(
      /*neg_a=*/false, a, /*neg_b=*/false, b,
      /*c_mod=*/(short)0, c, /*reuse_a=*/false, /*reuse_b=*/false);
}

// A fragment: 16x32 bf16 tile from row-major src (ld elements per row).
// Lane L<16: row=L, K chunks {0..7, 16..23}; lane>=16: row=L-16, K {8..15, 24..31}.
__device__ __forceinline__ v16bf load_a_frag(const unsigned short* base, int ld,
                                             int m_base, int k_base, int lane) {
  int row = m_base + (lane & 15);
  int k0  = k_base + ((lane & 16) ? 8 : 0);
  const unsigned short* p = base + row * ld + k0;
  Frag16 f;
  f.u[0] = *reinterpret_cast<const u32x4*>(p);
  f.u[1] = *reinterpret_cast<const u32x4*>(p + 16);
  return f.v;
}

// B fragment: 32x16 bf16, B[k][n] = src[n_base+n][k_base+k] (src row-major, so
// src rows act as columns of B -> computes A * src^T).
// Lane L<16: col=L, K {0..7, 16..23}; lane>=16: col=L-16, K {8..15, 24..31}.
__device__ __forceinline__ v16bf load_b_frag(const unsigned short* base, int ld,
                                             int n_base, int k_base, int lane) {
  int col = n_base + (lane & 15);
  int kk  = k_base + ((lane & 16) ? 8 : 0);
  const unsigned short* p = base + col * ld + kk;
  Frag16 f;
  f.u[0] = *reinterpret_cast<const u32x4*>(p);
  f.u[1] = *reinterpret_cast<const u32x4*>(p + 16);
  return f.v;
}

// ---------------- Kernel 1: f32 -> bf16 convert (8 elems/thread) ----------------
__global__ void __launch_bounds__(256)
cvt_bf16_kernel(const float* __restrict__ in, unsigned short* __restrict__ out, int n) {
  int i = (blockIdx.x * blockDim.x + threadIdx.x) * 8;
  if (i >= n) return;
  float4 a = *reinterpret_cast<const float4*>(in + i);
  float4 b = *reinterpret_cast<const float4*>(in + i + 4);
  union { unsigned short s[8]; u32x4 u; } r;
  r.s[0] = f32_to_bf16(a.x); r.s[1] = f32_to_bf16(a.y);
  r.s[2] = f32_to_bf16(a.z); r.s[3] = f32_to_bf16(a.w);
  r.s[4] = f32_to_bf16(b.x); r.s[5] = f32_to_bf16(b.y);
  r.s[6] = f32_to_bf16(b.z); r.s[7] = f32_to_bf16(b.w);
  *reinterpret_cast<u32x4*>(out + i) = r.u;
}

// ------- Kernel 2: W [K][N] f32 -> Wt [N][K] bf16, 64x64 LDS-tiled transpose ----
__global__ void __launch_bounds__(256)
transpose_w_kernel(const float* __restrict__ W, unsigned short* __restrict__ Wt) {
  __shared__ __align__(16) float tile[64][68];  // pad 68 -> 16B aligned rows, few conflicts
  int tn = blockIdx.x & 15;   // N tile of output (column tile of W)
  int tk = blockIdx.x >> 4;   // K tile (row tile of W)
  int r0 = tk * 64, c0 = tn * 64;
  int tid = threadIdx.x;

  // load 64x64 f32 tile, coalesced (float4 per lane, 4 iterations)
#pragma unroll
  for (int i = 0; i < 4; ++i) {
    int id = i * 256 + tid;      // 0..1023
    int r = id >> 4;             // 0..63
    int c4 = (id & 15) << 2;     // 0,4,...,60
    const float* gp = W + (size_t)(r0 + r) * D_MODEL + c0 + c4;
    *reinterpret_cast<float4*>(&tile[r][c4]) = *reinterpret_cast<const float4*>(gp);
  }
  __syncthreads();

  // write Wt[n][k] = W[k][n], coalesced along k
#pragma unroll
  for (int i = 0; i < 16; ++i) {
    int id = i * 256 + tid;      // 0..4095
    int cc = id >> 6;            // output row (n-local), 0..63
    int rr = id & 63;            // output col (k-local), contiguous across lanes
    Wt[(size_t)(c0 + cc) * D_MODEL + r0 + rr] = f32_to_bf16(tile[rr][cc]);
  }
}

// ---------------- Kernel 3: GEMM  C = A(bf16,[4096][1024]) * Wt^T + bias --------
// Each wave computes a 16x64 strip (4 WMMA accumulators, A-fragment reused 4x).
// mode 0: write bf16 head-split row-major  [b,h,s,dk]
// mode 1: write bf16 head-split transposed [b,h,dk,s]   (for V)
// mode 2: write f32 row-major [m][1024]                  (final output)
__global__ void __launch_bounds__(128)
proj_kernel(const unsigned short* __restrict__ A,
            const unsigned short* __restrict__ Bt,
            const float* __restrict__ bias,
            unsigned short* __restrict__ out_b16,
            float* __restrict__ out_f32, int mode) {
  int lane = threadIdx.x & 31;
  int wave = threadIdx.x >> 5;
  int tile = blockIdx.x * 4 + wave;  // 4096 wave-strips total
  int tm = tile >> 4;                // M tile (0..255)
  int tn4 = tile & 15;               // 64-wide N strip (0..15) == head index

  const unsigned short* arow = A + (size_t)(tm * 16 + (lane & 15)) * D_MODEL;
  const unsigned short* brow = Bt + (size_t)(tn4 * 64 + (lane & 15)) * D_MODEL;

  v8f acc[4];
#pragma unroll
  for (int t = 0; t < 4; ++t) acc[t] = v8f{0.f, 0.f, 0.f, 0.f, 0.f, 0.f, 0.f, 0.f};

  for (int k = 0; k < D_MODEL; k += 32) {
    // prefetch next iterations' operand lines (lowers to global_prefetch_b8)
    __builtin_prefetch(arow + k + 128, 0, 1);
    __builtin_prefetch(brow + k + 128, 0, 1);
    v16bf a = load_a_frag(A, D_MODEL, tm * 16, k, lane);
#pragma unroll
    for (int t = 0; t < 4; ++t) {
      v16bf b = load_b_frag(Bt, D_MODEL, tn4 * 64 + t * 16, k, lane);
      acc[t] = wmma_bf16(a, b, acc[t]);
    }
  }

  int h = tn4;
#pragma unroll
  for (int t = 0; t < 4; ++t) {
    int colg = tn4 * 64 + t * 16 + (lane & 15);
    int c = colg & 63;
    float bv = bias[colg];
#pragma unroll
    for (int v = 0; v < 8; ++v) {
      int row = tm * 16 + v + ((lane & 16) ? 8 : 0);  // global row in [0,4096)
      float val = acc[t][v] + bv;
      int b_ = row >> 11, s = row & (SEQ - 1);
      if (mode == 0) {
        out_b16[(((size_t)(b_ * NUM_HEADS + h) * SEQ) + s) * D_K + c] = f32_to_bf16(val);
      } else if (mode == 1) {
        out_b16[(((size_t)(b_ * NUM_HEADS + h) * D_K) + c) * SEQ + s] = f32_to_bf16(val);
      } else {
        out_f32[(size_t)row * D_MODEL + colg] = val;
      }
    }
  }
}

// ---------------- Kernel 4: attention ----------------
// One block (4 waves) per (b,h, 16-row q tile). Two-pass softmax, P in LDS,
// ctx = P * V via WMMA. Score matrix never touches HBM.
__global__ void __launch_bounds__(128)
attn_kernel(const unsigned short* __restrict__ q_ws,
            const unsigned short* __restrict__ k_ws,
            const unsigned short* __restrict__ vt_ws,
            unsigned short* __restrict__ ctx_ws,
            float* __restrict__ attn_out) {
  __shared__ __align__(16) unsigned short Pbuf[16 * SEQ];  // 64 KB
  __shared__ float stat_m[4][16];
  __shared__ float stat_l[4][16];
  __shared__ float fin_m[16];
  __shared__ float fin_l[16];

  int tid = threadIdx.x;
  int lane = tid & 31;
  int wave = tid >> 5;
  int bid = blockIdx.x;
  int qt = bid & 127;        // q tile within sequence
  int bh = bid >> 7;         // 0..31 = b*16 + h
  int b_ = bh >> 4, h = bh & 15;

  const unsigned short* qbase = q_ws + ((size_t)bh * SEQ + qt * 16) * D_K;
  const unsigned short* kbase = k_ws + (size_t)bh * SEQ * D_K;
  const unsigned short* vtbase = vt_ws + (size_t)bh * D_K * SEQ;

  // Q fragments for this 16-row tile (K-dim = 64 -> two fragments)
  v16bf qa0 = load_a_frag(qbase, D_K, 0, 0, lane);
  v16bf qa1 = load_a_frag(qbase, D_K, 0, 32, lane);

  // ---- pass 1: online max/sum over all 128 key tiles ----
  float m_run[8], l_run[8];
#pragma unroll
  for (int v = 0; v < 8; ++v) { m_run[v] = -3.0e38f; l_run[v] = 0.f; }

  for (int j = wave; j < SEQ / 16; j += 4) {
    // prefetch the K tile this wave will use next iteration
    __builtin_prefetch(kbase + (size_t)(j + 4) * 16 * D_K + (lane & 15) * D_K, 0, 1);
    v8f c = {0.f, 0.f, 0.f, 0.f, 0.f, 0.f, 0.f, 0.f};
    v16bf kb0 = load_b_frag(kbase, D_K, j * 16, 0, lane);
    v16bf kb1 = load_b_frag(kbase, D_K, j * 16, 32, lane);
    c = wmma_bf16(qa0, kb0, c);
    c = wmma_bf16(qa1, kb1, c);
#pragma unroll
    for (int v = 0; v < 8; ++v) {
      float x = c[v] * SCALE;
      float mo = m_run[v];
      float mn = fmaxf(mo, x);
      l_run[v] = l_run[v] * __expf(mo - mn) + __expf(x - mn);
      m_run[v] = mn;
    }
  }
  // reduce across the 16 lanes that share each row (xor within 16-lane group)
#pragma unroll
  for (int off = 1; off < 16; off <<= 1) {
#pragma unroll
    for (int v = 0; v < 8; ++v) {
      float mo = __shfl_xor(m_run[v], off, 32);
      float lo = __shfl_xor(l_run[v], off, 32);
      float mn = fmaxf(m_run[v], mo);
      l_run[v] = l_run[v] * __expf(m_run[v] - mn) + lo * __expf(mo - mn);
      m_run[v] = mn;
    }
  }
  if ((lane & 15) == 0) {
#pragma unroll
    for (int v = 0; v < 8; ++v) {
      int row = v + ((lane & 16) ? 8 : 0);
      stat_m[wave][row] = m_run[v];
      stat_l[wave][row] = l_run[v];
    }
  }
  __syncthreads();
  if (tid < 16) {
    float m = stat_m[0][tid], l = stat_l[0][tid];
#pragma unroll
    for (int w = 1; w < 4; ++w) {
      float mo = stat_m[w][tid], lo = stat_l[w][tid];
      float mn = fmaxf(m, mo);
      l = l * __expf(m - mn) + lo * __expf(mo - mn);
      m = mn;
    }
    fin_m[tid] = m;
    fin_l[tid] = l;
  }
  __syncthreads();

  float fm[8], fli[8];
#pragma unroll
  for (int v = 0; v < 8; ++v) {
    int r = v + ((lane & 16) ? 8 : 0);
    fm[v] = fin_m[r];
    fli[v] = 1.0f / fin_l[r];
  }

  // ---- pass 2: recompute scores, emit attn (f32 global) + P (bf16 LDS) ----
  for (int j = wave; j < SEQ / 16; j += 4) {
    v8f c = {0.f, 0.f, 0.f, 0.f, 0.f, 0.f, 0.f, 0.f};
    v16bf kb0 = load_b_frag(kbase, D_K, j * 16, 0, lane);
    v16bf kb1 = load_b_frag(kbase, D_K, j * 16, 32, lane);
    c = wmma_bf16(qa0, kb0, c);
    c = wmma_bf16(qa1, kb1, c);
    int col = j * 16 + (lane & 15);
#pragma unroll
    for (int v = 0; v < 8; ++v) {
      int row = v + ((lane & 16) ? 8 : 0);
      float p = __expf(c[v] * SCALE - fm[v]) * fli[v];
      attn_out[((size_t)bh * SEQ + qt * 16 + row) * SEQ + col] = p;
      Pbuf[row * SEQ + col] = f32_to_bf16(p);
    }
  }
  __syncthreads();

  // ---- ctx = P(16 x 2048) * V(2048 x 64); wave w -> dk columns [16w,16w+16) ----
  v8f acc = {0.f, 0.f, 0.f, 0.f, 0.f, 0.f, 0.f, 0.f};
  for (int kk = 0; kk < SEQ; kk += 32) {
    v16bf pa = load_a_frag(Pbuf, SEQ, 0, kk, lane);
    v16bf vb = load_b_frag(vtbase, SEQ, wave * 16, kk, lane);
    acc = wmma_bf16(pa, vb, acc);
  }
  int colc = h * D_K + wave * 16 + (lane & 15);
#pragma unroll
  for (int v = 0; v < 8; ++v) {
    int row = qt * 16 + v + ((lane & 16) ? 8 : 0);
    ctx_ws[((size_t)b_ * SEQ + row) * D_MODEL + colc] = f32_to_bf16(acc[v]);
  }
}

// ---------------- host ----------------
extern "C" void kernel_launch(void* const* d_in, const int* in_sizes, int n_in,
                              void* d_out, int out_size, void* d_ws, size_t ws_size,
                              hipStream_t stream) {
  (void)in_sizes; (void)n_in; (void)out_size; (void)ws_size;
  const float* Q_in = (const float*)d_in[0];
  const float* K_in = (const float*)d_in[1];
  const float* V_in = (const float*)d_in[2];
  const float* Wq = (const float*)d_in[3];
  const float* bq = (const float*)d_in[4];
  const float* Wk = (const float*)d_in[5];
  const float* bk = (const float*)d_in[6];
  const float* Wv = (const float*)d_in[7];
  const float* bv = (const float*)d_in[8];
  const float* Wfc = (const float*)d_in[9];
  const float* bfc = (const float*)d_in[10];

  float* out_f32 = (float*)d_out;                                  // [B,S,D]
  float* attn_out = out_f32 + (size_t)BATCH * SEQ * D_MODEL;       // [B,H,S,S]

  char* ws = (char*)d_ws;
  size_t off = 0;
  auto take = [&](size_t bytes) {
    void* p = ws + off;
    off += (bytes + 255) & ~(size_t)255;
    return p;
  };
  const size_t actBytes = (size_t)BATCH * SEQ * D_MODEL * 2;  // 8.4 MB bf16
  const size_t wBytes = (size_t)D_MODEL * D_MODEL * 2;        // 2.1 MB bf16
  unsigned short* xq = (unsigned short*)take(actBytes);
  unsigned short* xk = (unsigned short*)take(actBytes);
  unsigned short* xv = (unsigned short*)take(actBytes);
  unsigned short* wqt = (unsigned short*)take(wBytes);
  unsigned short* wkt = (unsigned short*)take(wBytes);
  unsigned short* wvt = (unsigned short*)take(wBytes);
  unsigned short* wfct = (unsigned short*)take(wBytes);
  unsigned short* q_ws = (unsigned short*)take(actBytes);   // [b,h,s,dk]
  unsigned short* k_ws = (unsigned short*)take(actBytes);   // [b,h,s,dk]
  unsigned short* vt_ws = (unsigned short*)take(actBytes);  // [b,h,dk,s]
  unsigned short* ctx_ws = (unsigned short*)take(actBytes); // [b,s,d]

  const int nAct = BATCH * SEQ * D_MODEL;  // 4,194,304
  cvt_bf16_kernel<<<nAct / (256 * 8), 256, 0, stream>>>(Q_in, xq, nAct);
  cvt_bf16_kernel<<<nAct / (256 * 8), 256, 0, stream>>>(K_in, xk, nAct);
  cvt_bf16_kernel<<<nAct / (256 * 8), 256, 0, stream>>>(V_in, xv, nAct);

  // 16x16 = 256 tiles of 64x64 per matrix
  transpose_w_kernel<<<256, 256, 0, stream>>>(Wq, wqt);
  transpose_w_kernel<<<256, 256, 0, stream>>>(Wk, wkt);
  transpose_w_kernel<<<256, 256, 0, stream>>>(Wv, wvt);
  transpose_w_kernel<<<256, 256, 0, stream>>>(Wfc, wfct);

  // 4096 wave-strips (16x64 each) / 4 waves per block
  proj_kernel<<<1024, 128, 0, stream>>>(xq, wqt, bq, q_ws, nullptr, 0);
  proj_kernel<<<1024, 128, 0, stream>>>(xk, wkt, bk, k_ws, nullptr, 0);
  proj_kernel<<<1024, 128, 0, stream>>>(xv, wvt, bv, vt_ws, nullptr, 1);

  // B*H*(S/16) = 4096 blocks
  attn_kernel<<<4096, 128, 0, stream>>>(q_ws, k_ws, vt_ws, ctx_ws, attn_out);

  proj_kernel<<<1024, 128, 0, stream>>>(ctx_ws, wfct, bfc, nullptr, out_f32, 2);
}
